// QuantSelfAttention_79182017069504
// MI455X (gfx1250) — compile-verified, matrix operand order checked by reading, not run on hardware
//
#include <hip/hip_runtime.h>
#include <hip/hip_bf16.h>

typedef __attribute__((ext_vector_type(16))) _Float16 v16h;
typedef __attribute__((ext_vector_type(8)))  _Float16 v8h;
typedef __attribute__((ext_vector_type(8)))  float    v8f;
typedef __attribute__((ext_vector_type(8)))  float    v8ff;
typedef __attribute__((ext_vector_type(4)))  float    v4f;
typedef __attribute__((ext_vector_type(4)))  int      v4i;

#define C_DIM   768
#define N_SEQ   1024
#define B_SZ    8
#define H_NUM   12
#define D_HEAD  64
#define M_TOT   (B_SZ * N_SEQ)          // 8192 rows
#define ELEMS_W (C_DIM * C_DIM)         // 589824
#define ELEMS_A (B_SZ * N_SEQ * C_DIM)  // 6291456

#if defined(__HIP_DEVICE_COMPILE__) && \
    __has_builtin(__builtin_amdgcn_global_load_async_to_lds_b128) && \
    __has_builtin(__builtin_amdgcn_s_wait_asynccnt)
#define USE_ASYNC_LDS 1
#endif

// exp(x)*2^24 LUT; idx = clip(round(16*(x-max)), -128, 0) + 128
__device__ __constant__ float g_lut[129] = {
    0,0,0,0,0,0,0,0, 0,0,0,0,0,0,0,0, 0,0,0,0,0,0,0,0, 0,0,0,0,0,0,0,0,
    0,0,0,0,0,0,0,0, 0,0,0,0,0,0,0,0, 0,0,0,0,0,0,0,0, 0,0,0,0,0,0,0,0,
    0,0,0,0,0,0,0,0, 0,0,0,0,0,0,0,0, 0,0,0,0,0,0,0,0, 0,0,0,0,0,0,0,0,
    0,0,0,0,0,0,0,0, 0,0,0,0,0,0,0,0,
    1.f,5.f,13.f,37.f,103.f,280.f,761.f,2070.f,5628.f,15298.f,41586.f,
    113043.f,307285.f,835288.f,2270549.f,6171992.f,16777216.f
};

// ---- WMMA fragment helpers (ISA 7.12.2, wave32) ----
// 16-bit A/B fragment: element e -> K = 16*(e>=8) + 8*(lane>=16) + e%8,
// i.e. two contiguous 8-halfword (16B) runs at +kb and +16+kb.
__device__ __forceinline__ v16h frag_from_f16(const _Float16* p, int lane) {
    const int kb = (lane & 16) >> 1;
    v8h lo = *(const v8h*)(p + kb);
    v8h hi = *(const v8h*)(p + 16 + kb);
    return __builtin_shufflevector(lo, hi, 0,1,2,3,4,5,6,7,8,9,10,11,12,13,14,15);
}
__device__ __forceinline__ v16h frag_from_f32(const float* p, int lane) {
    const int kb = (lane & 16) >> 1;
    v8ff lo = *(const v8ff*)(p + kb);
    v8ff hi = *(const v8ff*)(p + 16 + kb);
    v8h l = __builtin_convertvector(lo, v8h);
    v8h h = __builtin_convertvector(hi, v8h);
    return __builtin_shufflevector(l, h, 0,1,2,3,4,5,6,7,8,9,10,11,12,13,14,15);
}
__device__ __forceinline__ v16h frag_quant_f32(const float* p, int lane,
                                               float sw, float inv_sw, float lim) {
    const int kb = (lane & 16) >> 1;
    v8ff lo = *(const v8ff*)(p + kb);
    v8ff hi = *(const v8ff*)(p + 16 + kb);
#pragma unroll
    for (int e = 0; e < 8; ++e) {
        lo[e] = fminf(fmaxf(rintf(lo[e] * inv_sw), -lim), 127.f) * sw;
        hi[e] = fminf(fmaxf(rintf(hi[e] * inv_sw), -lim), 127.f) * sw;
    }
    v8h l = __builtin_convertvector(lo, v8h);
    v8h h = __builtin_convertvector(hi, v8h);
    return __builtin_shufflevector(l, h, 0,1,2,3,4,5,6,7,8,9,10,11,12,13,14,15);
}
// f32 C/D fragment: element i -> M = i + 8*(lane>=16), N = lane%16
__device__ __forceinline__ int cd_m(int lane, int i) {
    return i + ((lane & 16) >> 1);
}
__device__ __forceinline__ void atomic_max_absf(int* p, float v) {
    atomicMax(p, __float_as_int(fabsf(v)));   // valid ordering for non-negative floats
}

// -------------------- kernel 0: zero accumulators --------------------
__global__ void k_init(int* acc) {
    if (threadIdx.x < 16) acc[threadIdx.x] = 0;
}

// -------------------- kernel 1: weight max-abs (4 tensors, b128+ loads) ------------
__global__ void k_wmax(const float* w0, const float* w1, const float* w2, const float* w3,
                       int* acc) {
    const float* w = (blockIdx.y == 0) ? w0 : (blockIdx.y == 1) ? w1 :
                     (blockIdx.y == 2) ? w2 : w3;
    __shared__ float red[256];
    float m = 0.f;
    for (int i = blockIdx.x * 256 + threadIdx.x; i < ELEMS_W / 8; i += gridDim.x * 256) {
        v8ff v = ((const v8ff*)w)[i];
#pragma unroll
        for (int e = 0; e < 8; ++e) m = fmaxf(m, fabsf(v[e]));
    }
    red[threadIdx.x] = m;
    __syncthreads();
    for (int s = 128; s > 0; s >>= 1) {
        if (threadIdx.x < s) red[threadIdx.x] = fmaxf(red[threadIdx.x], red[threadIdx.x + s]);
        __syncthreads();
    }
    if (threadIdx.x == 0) atomic_max_absf(&acc[blockIdx.y], red[0]);
}

// -------------------- kernel 2: fused QKV projection (f16 WMMA) --------------------
// grid (512, 6, 3), block 256 (8 waves) -> 16x128 output tile per block
__global__ __launch_bounds__(256)
void k_qkv_gemm(const float* __restrict__ x,
                const float* __restrict__ Wq, const float* __restrict__ bq,
                const float* __restrict__ Wk, const float* __restrict__ bk,
                const float* __restrict__ Wv, const float* __restrict__ bv,
                _Float16* __restrict__ qkv_ws, int* __restrict__ acc) {
    const int z = blockIdx.z;
    const float* W = (z == 0) ? Wq : (z == 1) ? Wk : Wv;
    const float* bias = (z == 0) ? bq : (z == 1) ? bk : bv;
    _Float16* out = qkv_ws + (size_t)z * ELEMS_A;
    const float sw = __int_as_float(acc[z]) * (1.0f / 127.0f);
    const float inv_sw = (sw > 0.f) ? 1.0f / sw : 0.f;

    __shared__ float xs[16 * C_DIM];                      // 48 KB slab of x
    __shared__ float red[256];
    __shared__ __align__(16) _Float16 st[8][16][16];      // 4 KB per-wave store staging
    const int rowbase = blockIdx.x * 16;
#ifdef USE_ASYNC_LDS
    for (int i = threadIdx.x * 4; i < 16 * C_DIM; i += 256 * 4)
        __builtin_amdgcn_global_load_async_to_lds_b128(
            (v4i*)(x + (size_t)rowbase * C_DIM + i),
            (v4i*)(&xs[i]), 0, 0);
    __builtin_amdgcn_s_wait_asynccnt(0);
#else
    for (int i = threadIdx.x; i < 16 * C_DIM; i += 256)
        xs[i] = x[(size_t)rowbase * C_DIM + i];
#endif
    __syncthreads();

    const int wave = threadIdx.x >> 5;
    const int lane = threadIdx.x & 31;
    const int ntile = blockIdx.y * 8 + wave;  // 0..47
    const int ncol = ntile * 16 + (lane & 15);
    const float* wrow = W + (size_t)ncol * C_DIM;
    const float* arow = xs + (lane & 15) * C_DIM;

    v8f c = {};
    for (int kb = 0; kb < C_DIM / 32; ++kb) {
        __builtin_prefetch(wrow + kb * 32 + 64, 0, 1);
        v16h a = frag_from_f32(arow + kb * 32, lane);
        v16h bm = frag_quant_f32(wrow + kb * 32, lane, sw, inv_sw, 127.f);
        c = __builtin_amdgcn_wmma_f32_16x16x32_f16(false, a, false, bm,
                                                   (short)0, c, false, false);
    }
    const float bv_ = bias[ncol];
    float lmax = 0.f;
#pragma unroll
    for (int i = 0; i < 8; ++i) {
        int m = cd_m(lane, i);
        float v = c[i] + bv_;
        st[wave][m][lane & 15] = (_Float16)v;
        lmax = fmaxf(lmax, fabsf(v));
    }
    __builtin_amdgcn_wave_barrier();
    {   // re-read row-per-lane -> one b128 store per lane
        const int row = lane & 15, half = lane >> 4;
        v8h pack = *(const v8h*)&st[wave][row][half * 8];
        *(v8h*)&out[(size_t)(rowbase + row) * C_DIM + ntile * 16 + half * 8] = pack;
    }
    red[threadIdx.x] = lmax;
    __syncthreads();
    for (int s = 128; s > 0; s >>= 1) {
        if (threadIdx.x < s) red[threadIdx.x] = fmaxf(red[threadIdx.x], red[threadIdx.x + s]);
        __syncthreads();
    }
    if (threadIdx.x == 0) atomic_max_absf(&acc[4 + z], red[0]);
}

// -------------------- kernel 3: activation fake-quant (+ build V^T) --------------------
// grid (1024, 3); vectorized b128 in-place quant; z==2 scatters V into [b][h][d][key]
__global__ void k_aquant(_Float16* qkv_ws, _Float16* __restrict__ vT, const int* acc) {
    const int z = blockIdx.y;
    const float s = __int_as_float(acc[4 + z]) * (1.0f / 127.0f);
    const float inv = (s > 0.f) ? 1.0f / s : 0.f;
    v8h* buf = (v8h*)(qkv_ws + (size_t)z * ELEMS_A);
    for (int i = blockIdx.x * 256 + threadIdx.x; i < ELEMS_A / 8; i += gridDim.x * 256) {
        v8ff f = __builtin_convertvector(buf[i], v8ff);
#pragma unroll
        for (int e = 0; e < 8; ++e)
            f[e] = fminf(fmaxf(rintf(f[e] * inv), -128.f), 127.f) * s;
        v8h q = __builtin_convertvector(f, v8h);
        buf[i] = q;
        if (z == 2) {
            int i0 = i * 8;                         // 8 consecutive channels, same (b,n,h)
            int b = i0 / (N_SEQ * C_DIM);
            int rem = i0 % (N_SEQ * C_DIM);
            int n = rem / C_DIM, cc = rem % C_DIM;
            int h = cc >> 6, d0 = cc & 63;
            _Float16* vt = vT + (((size_t)b * H_NUM + h) * D_HEAD + d0) * N_SEQ + n;
#pragma unroll
            for (int e = 0; e < 8; ++e) vt[(size_t)e * N_SEQ] = q[e];
        }
    }
}

// -------------------- kernel 4: attention (WMMA scores + int softmax + WMMA P.V) ----
__global__ __launch_bounds__(256)
void k_attn(const _Float16* __restrict__ qkv_ws, const _Float16* __restrict__ vT,
            float* __restrict__ ctx) {
    extern __shared__ float S[];              // [16][1024] + 4*256 staging = 68 KB
    __shared__ float red[256], rowmax[16], rowsum[16];

    const int id = blockIdx.x;                // ((b*12 + h)*64 + qt)
    const int qt = id & 63;
    const int h = (id >> 6) % H_NUM;
    const int b = id / (64 * H_NUM);
    const int q0 = qt * 16;

    const _Float16* qws = qkv_ws;
    const _Float16* kws = qkv_ws + (size_t)ELEMS_A;

    const int wave = threadIdx.x >> 5;
    const int lane = threadIdx.x & 31;
    const size_t headoff = (size_t)b * N_SEQ * C_DIM + (size_t)h * D_HEAD;

    // ---- phase 1: S = (Q K^T) / sqrt(64) ----
    const _Float16* qp = qws + headoff + (size_t)(q0 + (lane & 15)) * C_DIM;
    v16h a0 = frag_from_f16(qp, lane);
    v16h a1 = frag_from_f16(qp + 32, lane);
    for (int kt = wave; kt < N_SEQ / 16; kt += 8) {
        const _Float16* kp = kws + headoff + (size_t)(kt * 16 + (lane & 15)) * C_DIM;
        v16h b0 = frag_from_f16(kp, lane);
        v16h b1 = frag_from_f16(kp + 32, lane);
        v8f c = {};
        c = __builtin_amdgcn_wmma_f32_16x16x32_f16(false, a0, false, b0, (short)0, c, false, false);
        c = __builtin_amdgcn_wmma_f32_16x16x32_f16(false, a1, false, b1, (short)0, c, false, false);
#pragma unroll
        for (int i = 0; i < 8; ++i)
            S[cd_m(lane, i) * N_SEQ + kt * 16 + (lane & 15)] = c[i] * 0.125f;
    }
    __syncthreads();

    // ---- phase 2: integer-LUT softmax, 16 rows x 1024, vectorized LDS ----
    const int row = threadIdx.x >> 4;
    const int seg = threadIdx.x & 15;
    const int base = row * N_SEQ + seg * 64;
    float lmax = -3.0e38f;
    for (int j = 0; j < 8; ++j) {
        v8ff v = *(const v8ff*)&S[base + j * 8];
#pragma unroll
        for (int e = 0; e < 8; ++e) lmax = fmaxf(lmax, v[e]);
    }
    red[threadIdx.x] = lmax;
    __syncthreads();
    if (seg == 0) {
        float m = red[row * 16];
        for (int t = 1; t < 16; ++t) m = fmaxf(m, red[row * 16 + t]);
        rowmax[row] = m;
    }
    __syncthreads();
    {
        float mx = rowmax[row], lsum = 0.f;
        for (int j = 0; j < 8; ++j) {
            v8ff v = *(const v8ff*)&S[base + j * 8];
#pragma unroll
            for (int e = 0; e < 8; ++e) {
                float d = (v[e] - mx) * 16.0f;
                int idx = (int)fminf(fmaxf(rintf(d), -128.f), 0.f) + 128;
                float y = g_lut[idx];
                v[e] = y;
                lsum += y;
            }
            *(v8ff*)&S[base + j * 8] = v;
        }
        red[threadIdx.x] = lsum;
    }
    __syncthreads();
    if (seg == 0) {
        float s = 0.f;
        for (int t = 0; t < 16; ++t) s += red[row * 16 + t];
        rowsum[row] = fmaxf(s, 1.0f);
    }
    __syncthreads();
    {
        float inv = 1.0f / rowsum[row];
        for (int j = 0; j < 8; ++j) {
            v8ff v = *(const v8ff*)&S[base + j * 8];
#pragma unroll
            for (int e = 0; e < 8; ++e) v[e] *= inv;
            *(v8ff*)&S[base + j * 8] = v;
        }
    }
    __syncthreads();

    // ---- phase 3: ctx = P (16x1024) x V (1024x64); waves 0..3 take d-tiles ----
    if (wave < 4) {
        const int d = wave * 16 + (lane & 15);
        const _Float16* vcol = vT + (((size_t)b * H_NUM + h) * D_HEAD + d) * N_SEQ;
        const float* prow = S + (lane & 15) * N_SEQ;
        v8f c = {};
        for (int kc = 0; kc < N_SEQ / 32; ++kc) {
            v16h a = frag_from_f32(prow + kc * 32, lane);
            v16h bm = frag_from_f16(vcol + kc * 32, lane);
            c = __builtin_amdgcn_wmma_f32_16x16x32_f16(false, a, false, bm,
                                                       (short)0, c, false, false);
        }
        // stage 16x16 f32 tile in LDS (region past S) -> coalesced wide stores
        float* stg = S + 16 * N_SEQ + wave * 256;
#pragma unroll
        for (int i = 0; i < 8; ++i)
            stg[cd_m(lane, i) * 16 + (lane & 15)] = c[i];
        __builtin_amdgcn_wave_barrier();
        const int r = lane & 15, half = lane >> 4;
        v8ff pack = *(const v8ff*)&stg[r * 16 + half * 8];
        *(v8ff*)&ctx[(size_t)b * N_SEQ * C_DIM + (size_t)(q0 + r) * C_DIM
                     + h * D_HEAD + wave * 16 + half * 8] = pack;
    }
}

// -------------------- kernel 5: mean-pool over sequence (float4) --------------------
__global__ void k_pool(const float* __restrict__ ctx, float* __restrict__ cbar) {
    int idx = blockIdx.x * 256 + threadIdx.x;       // 8 * 192 threads of float4
    if (idx >= B_SZ * C_DIM / 4) return;
    int b = idx / (C_DIM / 4), c4 = idx % (C_DIM / 4);
    v4f s = {};
    const v4f* p = (const v4f*)(ctx + (size_t)b * N_SEQ * C_DIM) + c4;
    for (int n = 0; n < N_SEQ; ++n) s += p[n * (C_DIM / 4)];
    ((v4f*)cbar)[idx] = s * (1.0f / N_SEQ);
}

// -------------------- kernel 6: output projection + bias + max-abs --------------------
__global__ void k_outproj(const float* __restrict__ cbar, const float* __restrict__ Wo,
                          const float* __restrict__ bo, float* __restrict__ raw, int* acc) {
    int idx = blockIdx.x * 256 + threadIdx.x;
    if (idx >= B_SZ * C_DIM) return;
    int b = idx / C_DIM, c = idx % C_DIM;
    float sw = __int_as_float(acc[3]) * (1.0f / 127.0f);
    float inv_sw = (sw > 0.f) ? 1.0f / sw : 0.f;
    float s = 0.f;
    for (int k = 0; k < C_DIM; k += 8) {
        v8ff wv = *(const v8ff*)(Wo + (size_t)c * C_DIM + k);
        v8ff xv = *(const v8ff*)(cbar + b * C_DIM + k);
#pragma unroll
        for (int e = 0; e < 8; ++e) {
            float qw = fminf(fmaxf(rintf(wv[e] * inv_sw), -127.f), 127.f) * sw;
            s += xv[e] * qw;
        }
    }
    s += bo[c];
    raw[idx] = s;
    atomic_max_absf(&acc[7], s);
}

// -------------------- kernel 7: final output fake-quant --------------------
__global__ void k_finalq(const float* __restrict__ raw, float* __restrict__ out,
                         const int* acc) {
    int idx = blockIdx.x * 256 + threadIdx.x;
    if (idx >= B_SZ * C_DIM) return;
    float s = __int_as_float(acc[7]) * (1.0f / 127.0f);
    float inv = (s > 0.f) ? 1.0f / s : 0.f;
    out[idx] = fminf(fmaxf(rintf(raw[idx] * inv), -128.f), 127.f) * s;
}

extern "C" void kernel_launch(void* const* d_in, const int* in_sizes, int n_in,
                              void* d_out, int out_size, void* d_ws, size_t ws_size,
                              hipStream_t stream) {
    const float* x  = (const float*)d_in[0];
    const float* Wq = (const float*)d_in[1];
    const float* bq = (const float*)d_in[2];
    const float* Wk = (const float*)d_in[3];
    const float* bk = (const float*)d_in[4];
    const float* Wv = (const float*)d_in[5];
    const float* bv = (const float*)d_in[6];
    const float* Wo = (const float*)d_in[7];
    const float* bo = (const float*)d_in[8];
    float* out = (float*)d_out;

    // workspace layout
    int* acc = (int*)d_ws;                                    // 16 ints (256B reserved)
    _Float16* qkv = (_Float16*)((char*)d_ws + 256);           // 3 * 6291456 halves
    _Float16* vT  = qkv + (size_t)3 * ELEMS_A;                // 6291456 halves
    float* ctx  = (float*)(vT + (size_t)ELEMS_A);             // 6291456 floats
    float* cbar = ctx + (size_t)ELEMS_A;                      // 6144
    float* raw  = cbar + B_SZ * C_DIM;                        // 6144

    k_init<<<1, 64, 0, stream>>>(acc);
    k_wmax<<<dim3(32, 4), 256, 0, stream>>>(Wq, Wk, Wv, Wo, acc);
    k_qkv_gemm<<<dim3(M_TOT / 16, 6, 3), 256, 0, stream>>>(x, Wq, bq, Wk, bk, Wv, bv, qkv, acc);
    k_aquant<<<dim3(1024, 3), 256, 0, stream>>>(qkv, vT, acc);
    k_attn<<<B_SZ * H_NUM * (N_SEQ / 16), 256,
             (16 * N_SEQ + 4 * 256) * sizeof(float), stream>>>(qkv, vT, ctx);
    k_pool<<<(B_SZ * C_DIM / 4 + 255) / 256, 256, 0, stream>>>(ctx, cbar);
    k_outproj<<<(B_SZ * C_DIM + 255) / 256, 256, 0, stream>>>(cbar, Wo, bo, raw, acc);
    k_finalq<<<(B_SZ * C_DIM + 255) / 256, 256, 0, stream>>>(raw, out, acc);
}
